// GateLoopBlock_764504179333
// MI455X (gfx1250) — matrix operand order repaired and checked by reading, not compiled, not run on hardware
//
#include <hip/hip_runtime.h>
#include <hip/hip_bf16.h>
#include <stdint.h>

// ---------------- problem constants (match reference) ----------------
#define BDIM 4
#define SLEN 4096
#define DDIM 1024
#define FDIM 4096
#define GDIM 32
#define TTOK (BDIM * SLEN)        // 16384 tokens
#define NCHUNK 16
#define CHUNK (SLEN / NCHUNK)     // 256
#define EPSV 1e-6f

typedef __bf16 bf16_t;
typedef __attribute__((ext_vector_type(16))) __bf16 v16bf;
typedef __attribute__((ext_vector_type(8)))  float  v8f;
typedef __attribute__((ext_vector_type(4)))  unsigned int u32x4;
typedef int i32x4v __attribute__((vector_size(16)));   // matches async-LDS builtin param

#define AS1 __attribute__((address_space(1)))
#define AS3 __attribute__((address_space(3)))

#ifndef __has_builtin
#define __has_builtin(x) 0
#endif
#if __has_builtin(__builtin_amdgcn_global_load_async_to_lds_b128) && \
    __has_builtin(__builtin_amdgcn_s_wait_asynccnt)
#define USE_ASYNC_LDS 1
#else
#define USE_ASYNC_LDS 0
#endif

__device__ __forceinline__ bf16_t f2bf(float f) {
  uint32_t u = __builtin_bit_cast(uint32_t, f);
  uint32_t r = (u + 0x7FFFu + ((u >> 16) & 1u)) >> 16;
  unsigned short hs = (unsigned short)r;
  return __builtin_bit_cast(bf16_t, hs);
}

// ---------------- conversion kernels ----------------
__global__ __launch_bounds__(256) void cvt_f32_to_bf16(const float* __restrict__ in,
                                                       bf16_t* __restrict__ out, size_t n) {
  size_t i = (size_t)blockIdx.x * 256 + threadIdx.x;
  if (i < n) out[i] = f2bf(in[i]);
}

// W [K,N] fp32 -> Wt [N,K] bf16 (so GEMM B-tiles are K-contiguous)
__global__ __launch_bounds__(256) void cvt_transpose_bf16(const float* __restrict__ W,
                                                          bf16_t* __restrict__ Wt, int K, int N) {
  size_t i = (size_t)blockIdx.x * 256 + threadIdx.x;
  if (i < (size_t)K * N) {
    int k = (int)(i / N);
    int n = (int)(i % N);
    Wt[(size_t)n * K + k] = f2bf(W[i]);
  }
}

// ---------------- WMMA bf16 GEMM ----------------
// C[M,N] = epi(A[M,K] @ Bt[N,K]^T + bias) (+ res)
// block tile 128x128, K-step 32, 256 threads = 8 waves, wave tile 32x64 (2x4 WMMA tiles)
// Double-buffered LDS; tiles stream in via global_load_async_to_lds_b128 (ASYNCcnt)
// while WMMAs consume the other buffer.
#define TBM 128
#define TBN 128
#define TBK 32
#define LSTR 40   // LDS row stride in halves (80B, 16B-aligned rows, padded vs. banks)

template <bool DO_GELU>
__global__ __launch_bounds__(256) void gemm_bf16_wmma(
    const bf16_t* __restrict__ A,   // [M,K] row-major bf16
    const bf16_t* __restrict__ Bt,  // [N,K] row-major bf16 (B transposed)
    const float*  __restrict__ bias,// [N] or null
    const float*  __restrict__ res, // [M,N] or null (residual add after activation)
    float*        __restrict__ Cf,  // [M,N] fp32 out or null
    bf16_t*       __restrict__ Cb,  // [M,N] bf16 out or null
    int M, int N, int K) {
  __shared__ bf16_t As[2][TBM * LSTR];
  __shared__ bf16_t Bs[2][TBN * LSTR];

  const int t    = threadIdx.x;
  const int lane = t & 31;          // wave32
  const int wave = t >> 5;          // 0..7
  const int l16  = lane & 15;
  const int hi   = lane >> 4;       // 0/1 half-wave
  const int wm   = (wave >> 1) * 32;  // wave row offset in block tile
  const int wn   = (wave & 1) * 64;   // wave col offset in block tile
  const int m0   = blockIdx.y * TBM;
  const int n0   = blockIdx.x * TBN;

  v8f acc[2][4];
  {
    v8f z = {0.f, 0.f, 0.f, 0.f, 0.f, 0.f, 0.f, 0.f};
    for (int i = 0; i < 2; ++i)
      for (int j = 0; j < 4; ++j) acc[i][j] = z;
  }

  const int lr = t >> 2;        // 0..63 : tile row loaded by this thread
  const int lc = (t & 3) * 8;   // half-offset within row (8 bf16 = 16B per move)

  // stage K-tile k0 into LDS buffer `buf` (async: no VGPR round-trip)
  auto stage = [&](int buf, int k0) {
    #pragma unroll
    for (int rep = 0; rep < TBM; rep += 64) {
      int row = rep + lr;
      const bf16_t* ga = A  + (size_t)(m0 + row) * K + k0 + lc;
      const bf16_t* gb = Bt + (size_t)(n0 + row) * K + k0 + lc;
#if USE_ASYNC_LDS
      __builtin_amdgcn_global_load_async_to_lds_b128(
          (AS1 i32x4v*)(bf16_t*)ga, (AS3 i32x4v*)&As[buf][row * LSTR + lc], 0, 0);
      __builtin_amdgcn_global_load_async_to_lds_b128(
          (AS1 i32x4v*)(bf16_t*)gb, (AS3 i32x4v*)&Bs[buf][row * LSTR + lc], 0, 0);
#else
      *(u32x4*)&As[buf][row * LSTR + lc] = *(const u32x4*)ga;
      *(u32x4*)&Bs[buf][row * LSTR + lc] = *(const u32x4*)gb;
#endif
    }
  };

  union Frag { u32x4 q[2]; v16bf v; };

  stage(0, 0);
#if USE_ASYNC_LDS
  __builtin_amdgcn_s_wait_asynccnt(0);
#endif
  __syncthreads();

  int cur = 0;
  for (int k0 = 0; k0 < K; k0 += TBK) {
    // ---- producer: stream next tile into the free buffer while we compute ----
    if (k0 + TBK < K) {
      stage(cur ^ 1, k0 + TBK);
      if (k0 + 2 * TBK < K) {  // hint tile-after-next toward L2 (global_prefetch_b8)
        __builtin_prefetch(A  + (size_t)(m0 + lr) * K + k0 + 2 * TBK + lc, 0, 2);
        __builtin_prefetch(Bt + (size_t)(n0 + lr) * K + k0 + 2 * TBK + lc, 0, 2);
      }
    }

    // ---- consumer: fragments per ISA 7.12.2 bf16 layouts ----
    // A 16x32: lanes 0-15 -> K {0..7, 16..23}; lanes 16-31 -> K {8..15, 24..31}
    Frag afr[2];
    #pragma unroll
    for (int mt = 0; mt < 2; ++mt) {
      const bf16_t* ap = &As[cur][(wm + mt * 16 + l16) * LSTR + hi * 8];
      afr[mt].q[0] = *(const u32x4*)(ap);        // k = kb + 0..7
      afr[mt].q[1] = *(const u32x4*)(ap + 16);   // k = kb + 16..23
    }
    // B 32x16: lanes 0-15 -> K {0..7, 16..23}; lanes 16-31 -> K {8..15, 24..31} (column = lane)
    Frag bfr[4];
    #pragma unroll
    for (int nt = 0; nt < 4; ++nt) {
      const bf16_t* bp = &Bs[cur][(wn + nt * 16 + l16) * LSTR + hi * 8];
      bfr[nt].q[0] = *(const u32x4*)(bp);
      bfr[nt].q[1] = *(const u32x4*)(bp + 16);
    }

    // ---- 8x v_wmma_f32_16x16x32_bf16 ----
    #pragma unroll
    for (int mt = 0; mt < 2; ++mt)
      #pragma unroll
      for (int nt = 0; nt < 4; ++nt)
        acc[mt][nt] = __builtin_amdgcn_wmma_f32_16x16x32_bf16(
            false, afr[mt].v, false, bfr[nt].v, (short)0, acc[mt][nt], false, false);

    // next iteration reads buffer cur^1: its async fills must be complete and
    // everyone must be done reading buffer cur before it is refilled.
#if USE_ASYNC_LDS
    __builtin_amdgcn_s_wait_asynccnt(0);
#endif
    __syncthreads();
    cur ^= 1;
  }

  // ---- epilogue: C/D layout = VGPR r -> M = r + 8*hi (+tile), N = l16 ----
  #pragma unroll
  for (int mt = 0; mt < 2; ++mt)
    #pragma unroll
    for (int nt = 0; nt < 4; ++nt)
      #pragma unroll
      for (int r = 0; r < 8; ++r) {
        int gm = m0 + wm + mt * 16 + hi * 8 + r;
        int gn = n0 + wn + nt * 16 + l16;
        float x = acc[mt][nt][r];
        if (bias) x += bias[gn];
        if (DO_GELU) {
          float x3 = x * x * x;
          x = 0.5f * x * (1.0f + tanhf(0.7978845608f * (x + 0.044715f * x3)));
        }
        if (res) x += res[(size_t)gm * N + gn];
        if (Cf) Cf[(size_t)gm * N + gn] = x;
        if (Cb) Cb[(size_t)gm * N + gn] = f2bf(x);
      }
}

// ---------------- gate prep: a_c = sigmoid(|a|) * e^{i*angle(a)}, kv = k*v ----------------
__global__ __launch_bounds__(256) void gate_prep(const float* __restrict__ a,
                                                 const float* __restrict__ k,
                                                 const float* __restrict__ v,
                                                 float* __restrict__ gr, float* __restrict__ gi,
                                                 float* __restrict__ kv, size_t n) {
  size_t i = (size_t)blockIdx.x * 256 + threadIdx.x;
  if (i >= n) return;
  size_t t = i / DDIM;
  int d = (int)(i % DDIM);
  float ar = a[t * 2 * DDIM + d];
  float ai = a[t * 2 * DDIM + DDIM + d];
  float mag = sqrtf(ar * ar + ai * ai);
  float s = 1.0f / (1.0f + expf(-mag));
  float inv = (mag > 0.0f) ? (1.0f / mag) : 0.0f;
  float cr = (mag > 0.0f) ? ar * inv : 1.0f;  // angle(0)=0 -> unit direction (1,0)
  float ci = ai * inv;
  gr[i] = s * cr;
  gi[i] = s * ci;
  kv[i] = k[i] * v[i];
}

// ---------------- chunked complex scan: h_t = a_t h_{t-1} + kv_t ----------------
__global__ __launch_bounds__(256) void scan_phase1(const float* __restrict__ gr,
                                                   const float* __restrict__ gi,
                                                   const float* __restrict__ kv,
                                                   float* __restrict__ cAr, float* __restrict__ cAi,
                                                   float* __restrict__ cHr, float* __restrict__ cHi) {
  int i = blockIdx.x * 256 + threadIdx.x;          // B*D*NCHUNK
  int d  = i % DDIM;
  int ch = (i / DDIM) % NCHUNK;
  int b  = i / (DDIM * NCHUNK);
  float Ar = 1.f, Ai = 0.f, Hr = 0.f, Hi = 0.f;
  int s0 = ch * CHUNK;
  for (int s = s0; s < s0 + CHUNK; ++s) {
    size_t off = ((size_t)b * SLEN + s) * DDIM + d;
    float ar = gr[off], ai = gi[off], x = kv[off];
    float nhr = ar * Hr - ai * Hi + x;
    float nhi = ar * Hi + ai * Hr;
    Hr = nhr; Hi = nhi;
    float nar = ar * Ar - ai * Ai;
    float nai = ar * Ai + ai * Ar;
    Ar = nar; Ai = nai;
  }
  size_t idx = ((size_t)b * DDIM + d) * NCHUNK + ch;
  cAr[idx] = Ar; cAi[idx] = Ai; cHr[idx] = Hr; cHi[idx] = Hi;
}

__global__ __launch_bounds__(256) void scan_phase2(const float* __restrict__ cAr,
                                                   const float* __restrict__ cAi,
                                                   const float* __restrict__ cHr,
                                                   const float* __restrict__ cHi,
                                                   float* __restrict__ hinr, float* __restrict__ hini) {
  int c = blockIdx.x * 256 + threadIdx.x;          // B*D channels
  size_t base = (size_t)c * NCHUNK;
  float hr = 0.f, hi = 0.f;
  for (int ch = 0; ch < NCHUNK; ++ch) {
    hinr[base + ch] = hr;
    hini[base + ch] = hi;
    float Ar = cAr[base + ch], Ai = cAi[base + ch];
    float nr = Ar * hr - Ai * hi + cHr[base + ch];
    float ni = Ar * hi + Ai * hr + cHi[base + ch];
    hr = nr; hi = ni;
  }
}

__global__ __launch_bounds__(256) void scan_phase3(const float* __restrict__ gr,
                                                   const float* __restrict__ gi,
                                                   const float* __restrict__ kv,
                                                   const float* __restrict__ q,
                                                   const float* __restrict__ hinr,
                                                   const float* __restrict__ hini,
                                                   float* __restrict__ yr, float* __restrict__ yi) {
  int i = blockIdx.x * 256 + threadIdx.x;
  int d  = i % DDIM;
  int ch = (i / DDIM) % NCHUNK;
  int b  = i / (DDIM * NCHUNK);
  size_t idx = ((size_t)b * DDIM + d) * NCHUNK + ch;
  float Hr = hinr[idx], Hi = hini[idx];
  int s0 = ch * CHUNK;
  for (int s = s0; s < s0 + CHUNK; ++s) {
    size_t off = ((size_t)b * SLEN + s) * DDIM + d;
    float ar = gr[off], ai = gi[off], x = kv[off];
    float nhr = ar * Hr - ai * Hi + x;
    float nhi = ar * Hi + ai * Hr;
    Hr = nhr; Hi = nhi;
    float qv = q[off];
    yr[off] = qv * Hr;
    yi[off] = qv * Hi;
  }
}

// ---------------- complex GroupNorm over (S, D/G) + silu(g) gating; emits Re only ----------------
__global__ __launch_bounds__(256) void groupnorm_gate(const float* __restrict__ yr,
                                                      const float* __restrict__ yi,
                                                      const float* __restrict__ g,
                                                      const float* __restrict__ gns,
                                                      const float* __restrict__ gnb,
                                                      float* __restrict__ yg,
                                                      bf16_t* __restrict__ ygb) {
  __shared__ float sr[256], si[256], s2[256];
  int b  = blockIdx.x / GDIM;
  int gr_ = blockIdx.x % GDIM;
  const int DC = DDIM / GDIM;           // 32
  const int NE = SLEN * DC;             // 131072
  float ar = 0.f, ai = 0.f, a2 = 0.f;
  for (int i = threadIdx.x; i < NE; i += 256) {
    int s = i / DC, dc = i % DC;
    size_t off = ((size_t)b * SLEN + s) * DDIM + gr_ * DC + dc;
    float r = yr[off], im = yi[off];
    ar += r; ai += im; a2 += r * r + im * im;
  }
  sr[threadIdx.x] = ar; si[threadIdx.x] = ai; s2[threadIdx.x] = a2;
  __syncthreads();
  for (int st = 128; st > 0; st >>= 1) {
    if (threadIdx.x < st) {
      sr[threadIdx.x] += sr[threadIdx.x + st];
      si[threadIdx.x] += si[threadIdx.x + st];
      s2[threadIdx.x] += s2[threadIdx.x + st];
    }
    __syncthreads();
  }
  float inv = 1.0f / (float)NE;
  float mr = sr[0] * inv, mi = si[0] * inv;
  float var = s2[0] * inv - (mr * mr + mi * mi);
  float rstd = rsqrtf(var + EPSV);
  for (int i = threadIdx.x; i < NE; i += 256) {
    int s = i / DC, dc = i % DC;
    int d = gr_ * DC + dc;
    size_t off = ((size_t)b * SLEN + s) * DDIM + d;
    float ynr = (yr[off] - mr) * rstd;
    float gg = g[off];
    float sg = gg / (1.0f + expf(-gg));        // silu(g), g purely real
    float val = (ynr * gns[d] + gnb[d]) * sg;  // Re(y_gated); Im unused downstream
    yg[off] = val;
    ygb[off] = f2bf(val);
  }
}

// ---------------- row LayerNorm (D=1024) with optional residual ----------------
__global__ __launch_bounds__(256) void layernorm_res(const float* __restrict__ X,
                                                     const float* __restrict__ R,
                                                     const float* __restrict__ sc,
                                                     const float* __restrict__ bi,
                                                     float* __restrict__ out,
                                                     bf16_t* __restrict__ outb) {
  __shared__ float s1[256], s2[256];
  size_t row = blockIdx.x;
  const float* xr = X + row * DDIM;
  const float* rr = R ? R + row * DDIM : nullptr;
  float vloc[4];
  float a1 = 0.f, a2 = 0.f;
  #pragma unroll
  for (int j = 0; j < 4; ++j) {
    int i = threadIdx.x + j * 256;
    float v = xr[i] + (rr ? rr[i] : 0.0f);
    vloc[j] = v; a1 += v; a2 += v * v;
  }
  s1[threadIdx.x] = a1; s2[threadIdx.x] = a2;
  __syncthreads();
  for (int st = 128; st > 0; st >>= 1) {
    if (threadIdx.x < st) {
      s1[threadIdx.x] += s1[threadIdx.x + st];
      s2[threadIdx.x] += s2[threadIdx.x + st];
    }
    __syncthreads();
  }
  float mu = s1[0] * (1.0f / DDIM);
  float var = s2[0] * (1.0f / DDIM) - mu * mu;
  float rstd = rsqrtf(var + EPSV);
  #pragma unroll
  for (int j = 0; j < 4; ++j) {
    int i = threadIdx.x + j * 256;
    float o = (vloc[j] - mu) * rstd * sc[i] + bi[i];
    out[row * DDIM + i] = o;
    if (outb) outb[row * DDIM + i] = f2bf(o);
  }
}

// ---------------- host orchestration ----------------
static inline size_t cdivz(size_t a, size_t b) { return (a + b - 1) / b; }

extern "C" void kernel_launch(void* const* d_in, const int* in_sizes, int n_in,
                              void* d_out, int out_size, void* d_ws, size_t ws_size,
                              hipStream_t stream) {
  (void)in_sizes; (void)n_in; (void)out_size; (void)ws_size;
  const float* x    = (const float*)d_in[0];
  const float* Wq   = (const float*)d_in[1];
  const float* Wk   = (const float*)d_in[2];
  const float* Wv   = (const float*)d_in[3];
  const float* Wa   = (const float*)d_in[4];
  const float* Wg   = (const float*)d_in[5];
  const float* Wo   = (const float*)d_in[6];
  const float* gns  = (const float*)d_in[7];
  const float* gnb  = (const float*)d_in[8];
  const float* ln1s = (const float*)d_in[9];
  const float* ln1b = (const float*)d_in[10];
  const float* W1   = (const float*)d_in[11];
  const float* b1   = (const float*)d_in[12];
  const float* W2   = (const float*)d_in[13];
  const float* b2   = (const float*)d_in[14];
  const float* ln2s = (const float*)d_in[15];
  const float* ln2b = (const float*)d_in[16];
  float* out = (float*)d_out;

  char* p = (char*)d_ws;
  auto take = [&](size_t bytes) -> void* {
    void* r = (void*)p;
    p += (bytes + 255) & ~(size_t)255;
    return r;
  };
  const size_t TD = (size_t)TTOK * DDIM;
  const size_t TF = (size_t)TTOK * FDIM;

  bf16_t* xb  = (bf16_t*)take(TD * 2);
  bf16_t* wqt = (bf16_t*)take((size_t)DDIM * DDIM * 2);
  bf16_t* wkt = (bf16_t*)take((size_t)DDIM * DDIM * 2);
  bf16_t* wvt = (bf16_t*)take((size_t)DDIM * DDIM * 2);
  bf16_t* wgt = (bf16_t*)take((size_t)DDIM * DDIM * 2);
  bf16_t* wot = (bf16_t*)take((size_t)DDIM * DDIM * 2);
  bf16_t* wat = (bf16_t*)take((size_t)DDIM * 2 * DDIM * 2);
  bf16_t* w1t = (bf16_t*)take((size_t)DDIM * FDIM * 2);
  bf16_t* w2t = (bf16_t*)take((size_t)FDIM * DDIM * 2);
  float* qb   = (float*)take(TD * 4);
  float* kb   = (float*)take(TD * 4);
  float* vb   = (float*)take(TD * 4);
  float* gm   = (float*)take(TD * 4);
  float* ab   = (float*)take(TD * 2 * 4);
  float* gr   = (float*)take(TD * 4);
  float* gi   = (float*)take(TD * 4);
  float* kv   = (float*)take(TD * 4);
  const size_t NCH_T = (size_t)BDIM * DDIM * NCHUNK;
  float* cAr  = (float*)take(NCH_T * 4);
  float* cAi  = (float*)take(NCH_T * 4);
  float* cHr  = (float*)take(NCH_T * 4);
  float* cHi  = (float*)take(NCH_T * 4);
  float* hinr = (float*)take(NCH_T * 4);
  float* hini = (float*)take(NCH_T * 4);
  float* yrb  = (float*)take(TD * 4);
  float* yib  = (float*)take(TD * 4);
  float* yg   = (float*)take(TD * 4);
  bf16_t* ygb = (bf16_t*)take(TD * 2);
  float* attn = (float*)take(TD * 4);
  float* x1   = (float*)take(TD * 4);
  bf16_t* x1b = (bf16_t*)take(TD * 2);
  bf16_t* hb  = (bf16_t*)take(TF * 2);
  float* mlp  = (float*)take(TD * 4);

  // --- precision prep ---
  cvt_f32_to_bf16<<<cdivz(TD, 256), 256, 0, stream>>>(x, xb, TD);
  cvt_transpose_bf16<<<cdivz((size_t)DDIM * DDIM, 256), 256, 0, stream>>>(Wq, wqt, DDIM, DDIM);
  cvt_transpose_bf16<<<cdivz((size_t)DDIM * DDIM, 256), 256, 0, stream>>>(Wk, wkt, DDIM, DDIM);
  cvt_transpose_bf16<<<cdivz((size_t)DDIM * DDIM, 256), 256, 0, stream>>>(Wv, wvt, DDIM, DDIM);
  cvt_transpose_bf16<<<cdivz((size_t)DDIM * DDIM, 256), 256, 0, stream>>>(Wg, wgt, DDIM, DDIM);
  cvt_transpose_bf16<<<cdivz((size_t)DDIM * DDIM, 256), 256, 0, stream>>>(Wo, wot, DDIM, DDIM);
  cvt_transpose_bf16<<<cdivz((size_t)DDIM * 2 * DDIM, 256), 256, 0, stream>>>(Wa, wat, DDIM, 2 * DDIM);
  cvt_transpose_bf16<<<cdivz((size_t)DDIM * FDIM, 256), 256, 0, stream>>>(W1, w1t, DDIM, FDIM);
  cvt_transpose_bf16<<<cdivz((size_t)FDIM * DDIM, 256), 256, 0, stream>>>(W2, w2t, FDIM, DDIM);

  // --- projections (WMMA bf16 GEMMs) ---
  dim3 gD(DDIM / TBN, TTOK / TBM);
  dim3 gA(2 * DDIM / TBN, TTOK / TBM);
  dim3 gF(FDIM / TBN, TTOK / TBM);
  gemm_bf16_wmma<false><<<gD, 256, 0, stream>>>(xb, wqt, nullptr, nullptr, qb, nullptr, TTOK, DDIM, DDIM);
  gemm_bf16_wmma<false><<<gD, 256, 0, stream>>>(xb, wkt, nullptr, nullptr, kb, nullptr, TTOK, DDIM, DDIM);
  gemm_bf16_wmma<false><<<gD, 256, 0, stream>>>(xb, wvt, nullptr, nullptr, vb, nullptr, TTOK, DDIM, DDIM);
  gemm_bf16_wmma<false><<<gD, 256, 0, stream>>>(xb, wgt, nullptr, nullptr, gm, nullptr, TTOK, DDIM, DDIM);
  gemm_bf16_wmma<false><<<gA, 256, 0, stream>>>(xb, wat, nullptr, nullptr, ab, nullptr, TTOK, 2 * DDIM, DDIM);

  // --- gate loop (chunked complex scan) ---
  gate_prep<<<cdivz(TD, 256), 256, 0, stream>>>(ab, kb, vb, gr, gi, kv, TD);
  scan_phase1<<<(BDIM * DDIM * NCHUNK) / 256, 256, 0, stream>>>(gr, gi, kv, cAr, cAi, cHr, cHi);
  scan_phase2<<<(BDIM * DDIM) / 256, 256, 0, stream>>>(cAr, cAi, cHr, cHi, hinr, hini);
  scan_phase3<<<(BDIM * DDIM * NCHUNK) / 256, 256, 0, stream>>>(gr, gi, kv, qb, hinr, hini, yrb, yib);

  // --- complex GroupNorm + silu(g) gating (real part only survives Wo) ---
  groupnorm_gate<<<BDIM * GDIM, 256, 0, stream>>>(yrb, yib, gm, gns, gnb, yg, ygb);

  // --- attn out = Re(y_gated) @ Wo ---
  gemm_bf16_wmma<false><<<gD, 256, 0, stream>>>(ygb, wot, nullptr, nullptr, attn, nullptr, TTOK, DDIM, DDIM);

  // --- x1 = LN1(x + attn) ---
  layernorm_res<<<TTOK, 256, 0, stream>>>(attn, x, ln1s, ln1b, x1, x1b);

  // --- MLP: h = gelu(x1@W1+b1) ; mlp = x1 + h@W2+b2 ---
  gemm_bf16_wmma<true ><<<gF, 256, 0, stream>>>(x1b, w1t, b1, nullptr, nullptr, hb, TTOK, FDIM, DDIM);
  gemm_bf16_wmma<false><<<gD, 256, 0, stream>>>(hb, w2t, b2, x1, mlp, nullptr, TTOK, DDIM, FDIM);

  // --- out = LN2(mlp) ---
  layernorm_res<<<TTOK, 256, 0, stream>>>(mlp, nullptr, ln2s, ln2b, out, nullptr);
}
